// RQLayer_53326313947283
// MI455X (gfx1250) — compile-verified, matrix operand order checked by reading, not run on hardware
//
#include <hip/hip_runtime.h>
#include <hip/hip_bf16.h>

// RQ-VAE forward for MI455X (gfx1250): 4 levels x (distance GEMM + argmin +
// residual update). Distance GEMM via v_wmma_f32_16x16x32_bf16 (f32 accum).
// M=32 rows per wave so each B (codebook) fragment loaded from LDS feeds two
// WMMAs -> halves LDS bandwidth per FLOP and interleaves two acc chains.
// x tile is staged into LDS with GLOBAL_LOAD_ASYNC_TO_LDS_B128 (ASYNCcnt).

#define NB   4
#define KC   256
#define DIM  128
#define BATCH 131072
#define ROWS_PER_WG 256
#define ROWS_PER_WAVE 32
#define CB_STRIDE 136   // bf16 elems per code row (272B: 16B aligned, bank-padded)
#define RES_STRIDE 132  // f32 elems per residual row (528B: 16B aligned, bank-padded)

typedef __attribute__((ext_vector_type(16))) __bf16         v16bf;
typedef __attribute__((ext_vector_type(8)))  float          v8f;
typedef __attribute__((ext_vector_type(16))) unsigned short v16us;
typedef __attribute__((ext_vector_type(4)))  int            v4i;

#if defined(__has_builtin)
#if __has_builtin(__builtin_amdgcn_global_load_async_to_lds_b128) && \
    __has_builtin(__builtin_amdgcn_s_wait_asynccnt)
#define HAVE_ASYNC_LDS 1
#endif
#endif

#if HAVE_ASYNC_LDS
typedef __attribute__((address_space(1))) v4i* gbl_v4i_p;   // global v4i*
typedef __attribute__((address_space(3))) v4i* lds_v4i_p;   // LDS v4i*
#endif

__device__ __forceinline__ unsigned short f2bf(float f) {
    unsigned int u = __builtin_bit_cast(unsigned int, f);
    unsigned int r = u + 0x7FFFu + ((u >> 16) & 1u);   // round-to-nearest-even
    return (unsigned short)(r >> 16);
}

struct U4Pair { uint4 a; uint4 b; };

__global__ __launch_bounds__(256, 1)
void rq_main(const float* __restrict__ x, const float* __restrict__ embeds,
             float* __restrict__ qout, int* __restrict__ codes_out,
             int* __restrict__ hist, float* __restrict__ lossAcc)
{
    __shared__ unsigned short sCB[KC * CB_STRIDE];            // 69,632 B
    __shared__ float          sRes[ROWS_PER_WG * RES_STRIDE]; // 135,168 B
    __shared__ float          sNorm[KC];
    __shared__ float          sLoss;

    const int tid  = threadIdx.x;
    const int lane = tid & 31;
    const int wave = tid >> 5;
    const int hi16 = lane >> 4;          // 0: lanes 0-15, 1: lanes 16-31
    const int l16  = lane & 15;
    const int rowBase = blockIdx.x * ROWS_PER_WG;  // global row of this WG
    const int wrow = wave * ROWS_PER_WAVE;         // wave's first local row

    // ---- load x tile into LDS residual buffer ----
#if HAVE_ASYNC_LDS
    for (int i = 0; i < ROWS_PER_WG / 8; ++i) {    // 256 rows * 32 float4/row
        int slot = tid + i * 256;
        int r = slot >> 5, c4 = slot & 31;
        const float* g = x + (size_t)(rowBase + r) * DIM + c4 * 4;
        float* l = &sRes[r * RES_STRIDE + c4 * 4];
        __builtin_amdgcn_global_load_async_to_lds_b128(
            (gbl_v4i_p)g, (lds_v4i_p)l, 0, 0);
    }
    __builtin_amdgcn_s_wait_asynccnt(0);
#else
    for (int i = 0; i < ROWS_PER_WG / 8; ++i) {
        int slot = tid + i * 256;
        int r = slot >> 5, c4 = slot & 31;
        float4 v = ((const float4*)x)[(size_t)(rowBase + r) * 32 + c4];
        float* dst = &sRes[r * RES_STRIDE + c4 * 4];
        dst[0] = v.x; dst[1] = v.y; dst[2] = v.z; dst[3] = v.w;
    }
#endif
    if (tid == 0) sLoss = 0.0f;
    __syncthreads();

    float lloss = 0.0f;

    for (int level = 0; level < NB; ++level) {
        // ---- stage codebook level as bf16 in LDS + per-code squared norm ----
        {
            const float* src = embeds + ((size_t)(level * KC + tid)) * DIM;
            unsigned short* dst = &sCB[tid * CB_STRIDE];
            float nrm = 0.0f;
            #pragma unroll
            for (int d = 0; d < DIM; d += 4) {
                float4 v = *(const float4*)(src + d);
                nrm += v.x*v.x + v.y*v.y + v.z*v.z + v.w*v.w;
                dst[d+0] = f2bf(v.x); dst[d+1] = f2bf(v.y);
                dst[d+2] = f2bf(v.z); dst[d+3] = f2bf(v.w);
            }
            sNorm[tid] = nrm;
        }
        __syncthreads();

        // ---- build A fragments for 2 row-blocks (ISA 7.12.2 A layout) ----
        // lane<16: row M=l16, K = s*32+{0..7,16..23}; lane>=16: K = s*32+{8..15,24..31}
        v16bf afrag[2][4];
        #pragma unroll
        for (int b = 0; b < 2; ++b) {
            const float* rp = &sRes[(wrow + b * 16 + l16) * RES_STRIDE];
            #pragma unroll
            for (int s = 0; s < 4; ++s) {
                const int kb = s * 32 + (hi16 ? 8 : 0);
                v16us u;
                #pragma unroll
                for (int j = 0; j < 8; ++j) {
                    u[j]     = f2bf(rp[kb + j]);
                    u[j + 8] = f2bf(rp[kb + 16 + j]);
                }
                afrag[b][s] = __builtin_bit_cast(v16bf, u);
            }
        }

        // ---- distance GEMM + running argmin over 16 N-tiles of 16 codes ----
        float best[2][8]; int bidx[2][8];
        #pragma unroll
        for (int r = 0; r < 8; ++r) {
            best[0][r] = 3.4e38f; bidx[0][r] = 0;
            best[1][r] = 3.4e38f; bidx[1][r] = 0;
        }

        for (int t = 0; t < 16; ++t) {
            const int col = t * 16 + l16;      // this lane's code column
            v8f acc0 = {}, acc1 = {};
            #pragma unroll
            for (int s = 0; s < 4; ++s) {
                const int kb = s * 32 + (hi16 ? 8 : 0);
                const unsigned short* bp = &sCB[col * CB_STRIDE + kb];
                U4Pair pr;
                pr.a = *(const uint4*)(bp);        // K = kb..kb+7
                pr.b = *(const uint4*)(bp + 16);   // K = kb+16..kb+23
                v16bf bfrag = __builtin_bit_cast(v16bf, pr);
                acc0 = __builtin_amdgcn_wmma_f32_16x16x32_bf16(
                         false, afrag[0][s], false, bfrag, (short)0, acc0, false, false);
                acc1 = __builtin_amdgcn_wmma_f32_16x16x32_bf16(
                         false, afrag[1][s], false, bfrag, (short)0, acc1, false, false);
            }
            const float en = sNorm[col];
            #pragma unroll
            for (int r = 0; r < 8; ++r) {        // score = ||e||^2 - 2 x.e
                float s0 = en - 2.0f * acc0[r];
                if (s0 < best[0][r]) { best[0][r] = s0; bidx[0][r] = col; }
                float s1 = en - 2.0f * acc1[r];
                if (s1 < best[1][r]) { best[1][r] = s1; bidx[1][r] = col; }
            }
        }

        // ---- argmin reduce across the 16 lanes of each half (rows split) ----
        #pragma unroll
        for (int m = 1; m < 16; m <<= 1) {
            #pragma unroll
            for (int b = 0; b < 2; ++b) {
                #pragma unroll
                for (int r = 0; r < 8; ++r) {
                    float os = __shfl_xor(best[b][r], m, 32);
                    int   oi = __shfl_xor(bidx[b][r], m, 32);
                    if (os < best[b][r] || (os == best[b][r] && oi < bidx[b][r])) {
                        best[b][r] = os; bidx[b][r] = oi;
                    }
                }
            }
        }

        // ---- emit codes + histogram (lane 0: rows 0-7, lane 16: rows 8-15) ----
        if (l16 == 0) {
            #pragma unroll
            for (int b = 0; b < 2; ++b) {
                const int rb = rowBase + wrow + b * 16 + (hi16 ? 8 : 0);
                #pragma unroll
                for (int r = 0; r < 8; ++r) {
                    codes_out[(size_t)(rb + r) * NB + level] = bidx[b][r];
                    atomicAdd(&hist[level * KC + bidx[b][r]], 1);
                }
            }
        }

        // broadcast chosen code per local row (0..31) to all lanes
        int idxRow[32];
        #pragma unroll
        for (int r = 0; r < 8; ++r) {
            idxRow[r]      = __shfl(bidx[0][r],  0, 32);
            idxRow[r + 8]  = __shfl(bidx[0][r], 16, 32);
            idxRow[r + 16] = __shfl(bidx[1][r],  0, 32);
            idxRow[r + 24] = __shfl(bidx[1][r], 16, 32);
        }

        if (level + 1 < NB)  // warm next level's codebook into cache
            __builtin_prefetch(embeds + (size_t)((level + 1) * KC) * DIM, 0, 0);

        // ---- residual update (f32 codebook from global/L2) + loss accum ----
        #pragma unroll
        for (int q = 0; q < ROWS_PER_WAVE; ++q) {
            const int code = idxRow[q];
            const float4 xq = *(const float4*)
                (embeds + ((size_t)(level * KC + code)) * DIM + lane * 4);
            float* rpq = &sRes[(wrow + q) * RES_STRIDE + lane * 4];
            float4 rv = *(const float4*)rpq;
            rv.x -= xq.x; rv.y -= xq.y; rv.z -= xq.z; rv.w -= xq.w;
            *(float4*)rpq = rv;
            lloss += rv.x*rv.x + rv.y*rv.y + rv.z*rv.z + rv.w*rv.w;
        }
        __syncthreads();   // codebook buffer reuse + residual visibility
    }

    // ---- per-WG loss reduction ----
    atomicAdd(&sLoss, lloss);
    __syncthreads();
    if (tid == 0) atomicAdd(lossAcc, sLoss);

    // ---- quantized_x = x_orig - final residual ----
    for (int i = 0; i < ROWS_PER_WG / 8; ++i) {
        int slot = tid + i * 256;
        int r = slot >> 5, c4 = slot & 31;
        float4 xv = ((const float4*)x)[(size_t)(rowBase + r) * 32 + c4];
        const float* rq = &sRes[r * RES_STRIDE + c4 * 4];
        float4 qv;
        qv.x = xv.x - rq[0]; qv.y = xv.y - rq[1];
        qv.z = xv.z - rq[2]; qv.w = xv.w - rq[3];
        ((float4*)qout)[(size_t)(rowBase + r) * 32 + c4] = qv;
    }
}

__global__ void rq_init(int* hist, float* lossAcc) {
    int i = blockIdx.x * blockDim.x + threadIdx.x;
    if (i < NB * KC) hist[i] = 0;
    if (i == 0) *lossAcc = 0.0f;
}

__global__ void rq_final(const int* __restrict__ hist, const float* __restrict__ lossAcc,
                         float* __restrict__ out_loss, int* __restrict__ out_unused) {
    __shared__ int cnt;
    if (threadIdx.x == 0) cnt = 0;
    __syncthreads();
    int c = 0;
    for (int i = threadIdx.x; i < NB * KC; i += blockDim.x) c += (hist[i] == 0);
    atomicAdd(&cnt, c);
    __syncthreads();
    if (threadIdx.x == 0) {
        // sum over levels of (1+beta)*mean(res^2), divided by NB
        *out_loss = *lossAcc * ((1.0f + 0.25f) / (float)NB / ((float)BATCH * (float)DIM));
        *out_unused = cnt;
    }
}

extern "C" void kernel_launch(void* const* d_in, const int* in_sizes, int n_in,
                              void* d_out, int out_size, void* d_ws, size_t ws_size,
                              hipStream_t stream) {
    const float* x      = (const float*)d_in[0];   // [131072,128] f32
    const float* embeds = (const float*)d_in[1];   // [4,256,128]  f32

    float* out      = (float*)d_out;
    float* qout     = out;                               // 16,777,216 f32
    float* out_loss = out + (size_t)BATCH * DIM;         // 1
    int*   out_unus = (int*)(out + (size_t)BATCH * DIM + 1);
    int*   codes    = (int*)(out + (size_t)BATCH * DIM + 2); // [131072,4] i32

    int*   hist    = (int*)d_ws;                          // 1024 ints
    float* lossAcc = (float*)((char*)d_ws + NB * KC * sizeof(int));

    rq_init<<<4, 256, 0, stream>>>(hist, lossAcc);
    rq_main<<<BATCH / ROWS_PER_WG, 256, 0, stream>>>(x, embeds, qout, codes, hist, lossAcc);
    rq_final<<<1, 256, 0, stream>>>(hist, lossAcc, out_loss, out_unus);
}
